// VINR_9448928051384
// MI455X (gfx1250) — compile-verified, matrix operand order checked by reading, not run on hardware
//
#include <hip/hip_runtime.h>
#include <cmath>

typedef __attribute__((ext_vector_type(16))) _Float16 v16h;
typedef __attribute__((ext_vector_type(4)))  _Float16 v4h;
typedef __attribute__((ext_vector_type(8)))  float    v8f;

#define Bb 4
#define Cc 3
#define Fn 4
#define Hh 512
#define Wd 512
#define HWp ((size_t)Hh * Wd)

// Build a 16-bit WMMA operand (A: 16x32, or B: 32x16) for this lane from a
// canonical row of 32 contiguous f16 K-values in LDS.
// Layout (ISA 7.12.2, 16-bit 16x32): half j -> v=j>>1, p=j&1,
//   k = (v>=4 ? 16 : 0) + 8*g + 2*(v&3) + p     (g = lane>>4)
// => j 0..3  : k = 8g+0..3    ; j 4..7  : k = 8g+4..7
//    j 8..11 : k = 16+8g+0..3 ; j 12..15: k = 16+8g+4..7
__device__ __forceinline__ v16h load_op16(const _Float16* __restrict__ row, int g) {
    const _Float16* p0 = row + g * 8;
    union { v16h v; v4h q[4]; } u;
    u.q[0] = *(const v4h*)(p0);
    u.q[1] = *(const v4h*)(p0 + 4);
    u.q[2] = *(const v4h*)(p0 + 16);
    u.q[3] = *(const v4h*)(p0 + 20);
    return u.v;
}

__global__ __launch_bounds__(128) void vinr_fused(
    const float* __restrict__ frames, const float* __restrict__ tvec,
    const float* __restrict__ We, const float* __restrict__ Wm,
    const float* __restrict__ Wg, const float* __restrict__ bg,
    float* __restrict__ out)
{
    // Canonical f16 staging: [row][k], row strides chosen so every v4h (8B)
    // access stays 8B-aligned (72B and 136B row pitches).
    __shared__ __align__(16) _Float16 lds_we[64][36];  // We rows, k 0..11 real
    __shared__ __align__(16) _Float16 lds_wm[64][68];  // Wm rows, k 0..63
    __shared__ __align__(16) _Float16 lds_wg[16][68];  // Wg rows 0..11 real
    __shared__ __align__(16) _Float16 lds_fr[64][36];  // frames [pixel][k<12]
    __shared__ __align__(16) _Float16 lds_nm[64][68];  // normalized [pixel][c]
    __shared__ __align__(16) _Float16 lds_md[64][68];  // mod        [pixel][c]
    __shared__ __align__(16) float    lds_fv[64][16];  // [pixel][flow0..7,vis0..3]

    const int tid  = threadIdx.x;
    const int lane = tid & 31;
    const int wave = tid >> 5;
    const int lm   = lane & 15;   // M row (A) / N column (B) within tile
    const int g    = lane >> 4;   // half-wave group

    const int b = blockIdx.x / Hh;
    const int h = blockIdx.x % Hh;

    // ---- stage all weights to LDS once (f16, zero-padded K) ----
    for (int idx = tid; idx < 64 * 12; idx += 128) {
        int e = idx / 12, k = idx - e * 12;
        lds_we[e][k] = (_Float16)We[idx];
    }
    for (int idx = tid; idx < 64 * 20; idx += 128) {
        int e = idx / 20, k = 12 + (idx - e * 20);
        lds_we[e][k] = (_Float16)0.f;
    }
    for (int idx = tid; idx < 64 * 64; idx += 128)
        lds_wm[idx >> 6][idx & 63] = (_Float16)Wm[idx];
    for (int idx = tid; idx < 16 * 64; idx += 128) {
        int o = idx >> 6, k = idx & 63;
        lds_wg[o][k] = (o < 12) ? (_Float16)Wg[o * 64 + k] : (_Float16)0.f;
    }
    for (int idx = tid; idx < 64 * 20; idx += 128) {  // frame K padding 12..31
        int p = idx / 20, k = 12 + (idx - p * 20);
        lds_fr[p][k] = (_Float16)0.f;
    }
    __syncthreads();

    // ---- per-lane A-matrix operands (weights, resident in VGPRs) ----
    v16h a_we[4];          // enc channel tiles, K=32 (12 real)
    v16h a_wm[4][2];       // mod channel tiles x K-chunks
    v16h a_wg[2];          // fv output tile x K-chunks
#pragma unroll
    for (int t = 0; t < 4; ++t) a_we[t] = load_op16(&lds_we[t * 16 + lm][0], g);
#pragma unroll
    for (int t = 0; t < 4; ++t)
#pragma unroll
        for (int kc = 0; kc < 2; ++kc)
            a_wm[t][kc] = load_op16(&lds_wm[t * 16 + lm][kc * 32], g);
#pragma unroll
    for (int kc = 0; kc < 2; ++kc) a_wg[kc] = load_op16(&lds_wg[lm][kc * 32], g);

    const float tb = tvec[b];
    const int   p  = wave * 16 + lm;  // this lane's pixel within the 64-chunk

    const size_t warp_base = (size_t)Bb * Cc * HWp;
    const size_t vis_base  = warp_base + (size_t)Bb * Cc * Fn * HWp;

#pragma unroll 1
    for (int iter = 0; iter < 8; ++iter) {
        const int px0 = iter * 64;
        __syncthreads();  // previous chunk's warp phase done before reuse

        // ---- stage frame chunk: [64 px][k<12], coalesced 64-lane row reads ----
        for (int idx = tid; idx < 12 * 64; idx += 128) {
            int k = idx >> 6, pp = idx & 63;
            lds_fr[pp][k] =
                (_Float16)frames[((size_t)b * 12 + k) * HWp + (size_t)h * Wd + px0 + pp];
        }
        if (iter < 7) {  // async-style prefetch of next chunk -> global_prefetch_b8
            for (int k = tid; k < 12; k += 128)
                __builtin_prefetch(
                    &frames[((size_t)b * 12 + k) * HWp + (size_t)h * Wd + px0 + 64], 0, 0);
        }
        __syncthreads();

        // ---- stage 1: enc[64ch x 16px] = We @ frames ----
        v8f enc[4];
        {
            v16h b_fr = load_op16(&lds_fr[p][0], g);
            v8f  z    = {};
#pragma unroll
            for (int t = 0; t < 4; ++t)
                enc[t] = __builtin_amdgcn_wmma_f32_16x16x32_f16(
                    false, a_we[t], false, b_fr, (short)0, z, false, false);
        }
        // per-pixel min/max over 64 channels: in-lane over 4x8 accs + xor16 merge
        float mn = 1e30f, mx = -1e30f;
#pragma unroll
        for (int t = 0; t < 4; ++t)
#pragma unroll
            for (int r = 0; r < 8; ++r) {
                float v = enc[t][r];
                mn = fminf(mn, v);
                mx = fmaxf(mx, v);
            }
        mn = fminf(mn, __shfl_xor(mn, 16, 32));
        mx = fmaxf(mx, __shfl_xor(mx, 16, 32));
        const float sc = 2.f / (mx - mn);
#pragma unroll
        for (int t = 0; t < 4; ++t) {
            union { v4h q; _Float16 hh[4]; } q0, q1;
#pragma unroll
            for (int r = 0; r < 4; ++r) {
                q0.hh[r] = (_Float16)((enc[t][r] - mn) * sc - 1.f);
                q1.hh[r] = (_Float16)((enc[t][4 + r] - mn) * sc - 1.f);
            }
            const int c0 = t * 16 + g * 8;  // D row r -> channel t*16 + 8g + r
            *(v4h*)&lds_nm[p][c0]     = q0.q;
            *(v4h*)&lds_nm[p][c0 + 4] = q1.q;
        }
        __syncthreads();

        // ---- stage 2: mod[64 x 16px] = Wm @ normalized (K=64 in 2 chunks) ----
        v8f md[4];
#pragma unroll
        for (int t = 0; t < 4; ++t) md[t] = (v8f){};
#pragma unroll
        for (int kc = 0; kc < 2; ++kc) {
            v16h bn = load_op16(&lds_nm[p][kc * 32], g);
#pragma unroll
            for (int t = 0; t < 4; ++t)
                md[t] = __builtin_amdgcn_wmma_f32_16x16x32_f16(
                    false, a_wm[t][kc], false, bn, (short)0, md[t], false, false);
        }
#pragma unroll
        for (int t = 0; t < 4; ++t) {
            union { v4h q; _Float16 hh[4]; } q0, q1;
#pragma unroll
            for (int r = 0; r < 4; ++r) {
                q0.hh[r] = (_Float16)md[t][r];
                q1.hh[r] = (_Float16)md[t][4 + r];
            }
            const int c0 = t * 16 + g * 8;
            *(v4h*)&lds_md[p][c0]     = q0.q;
            *(v4h*)&lds_md[p][c0 + 4] = q1.q;
        }
        __syncthreads();

        // ---- stage 3: fv[12 x 16px] = Wg @ mod ----
        v8f fv = {};
        {
            v16h bm0 = load_op16(&lds_md[p][0],  g);
            v16h bm1 = load_op16(&lds_md[p][32], g);
            fv = __builtin_amdgcn_wmma_f32_16x16x32_f16(
                false, a_wg[0], false, bm0, (short)0, fv, false, false);
            fv = __builtin_amdgcn_wmma_f32_16x16x32_f16(
                false, a_wg[1], false, bm1, (short)0, fv, false, false);
        }
        const int w = px0 + p;
        if (g == 0) {
            // rows 0..7 = all 8 flow components of pixel p
            float4 fA, fB;
            fA.x = fv[0] + tb * bg[0]; fA.y = fv[1] + tb * bg[1];
            fA.z = fv[2] + tb * bg[2]; fA.w = fv[3] + tb * bg[3];
            fB.x = fv[4] + tb * bg[4]; fB.y = fv[5] + tb * bg[5];
            fB.z = fv[6] + tb * bg[6]; fB.w = fv[7] + tb * bg[7];
            *(float4*)&lds_fv[p][0] = fA;
            *(float4*)&lds_fv[p][4] = fB;
        } else {
            // rows 0..3 = vis logits (o = 8..11): in-lane softmax
            float l0 = fv[0] + tb * bg[8],  l1 = fv[1] + tb * bg[9];
            float l2 = fv[2] + tb * bg[10], l3 = fv[3] + tb * bg[11];
            float m  = fmaxf(fmaxf(l0, l1), fmaxf(l2, l3));
            float e0 = __expf(l0 - m), e1 = __expf(l1 - m);
            float e2 = __expf(l2 - m), e3 = __expf(l3 - m);
            float inv = 1.f / (e0 + e1 + e2 + e3);
            float4 vv; vv.x = e0 * inv; vv.y = e1 * inv; vv.z = e2 * inv; vv.w = e3 * inv;
            *(float4*)&lds_fv[p][8] = vv;
            *(float4*)(out + vis_base + ((size_t)(b * Hh + h) * Wd + w) * 4) = vv;
        }
        __syncthreads();

        // ---- warping: (pixel,frame) pairs, 4 consecutive lanes = 1 pixel ----
#pragma unroll
        for (int it2 = 0; it2 < 2; ++it2) {
            const int pair = it2 * 128 + tid;   // 0..255 = 64 px * 4 frames
            const int pp = pair >> 2;
            const int f  = pair & 3;
            const int ww = px0 + pp;
            const float fx  = lds_fv[pp][2 * f];
            const float fy  = lds_fv[pp][2 * f + 1];
            const float vpf = lds_fv[pp][8 + f];
            const float sx = (float)ww + fx;
            const float sy = (float)h + fy;
            const float x0f = floorf(sx), y0f = floorf(sy);
            const int ix0 = (int)x0f, iy0 = (int)y0f;
            const int ix1 = ix0 + 1,  iy1 = iy0 + 1;
            const float wx1 = sx - x0f, wx0 = 1.f - wx1;
            const float wy1 = sy - y0f, wy0 = 1.f - wy1;
            const float mx0 = (ix0 >= 0 && ix0 < Wd) ? 1.f : 0.f;
            const float mx1 = (ix1 >= 0 && ix1 < Wd) ? 1.f : 0.f;
            const float my0 = (iy0 >= 0 && iy0 < Hh) ? 1.f : 0.f;
            const float my1 = (iy1 >= 0 && iy1 < Hh) ? 1.f : 0.f;
            const int cx0 = min(max(ix0, 0), Wd - 1), cx1 = min(max(ix1, 0), Wd - 1);
            const int cy0 = min(max(iy0, 0), Hh - 1), cy1 = min(max(iy1, 0), Hh - 1);
#pragma unroll
            for (int c = 0; c < 3; ++c) {
                const float* pl = frames + ((size_t)b * 12 + c * 4 + f) * HWp;
                const float t00 = pl[(size_t)cy0 * Wd + cx0] * (mx0 * my0);
                const float t10 = pl[(size_t)cy0 * Wd + cx1] * (mx1 * my0);
                const float t01 = pl[(size_t)cy1 * Wd + cx0] * (mx0 * my1);
                const float t11 = pl[(size_t)cy1 * Wd + cx1] * (mx1 * my1);
                const float val = t00 * (wx0 * wy0) + t10 * (wx1 * wy0) +
                                  t01 * (wx0 * wy1) + t11 * (wx1 * wy1);
                // warped: (B, C, F, H, W)
                out[warp_base + (((size_t)(b * Cc + c) * Fn + f) * Hh + h) * Wd + ww] = val;
                // masked: sum over f via 2x shfl_xor within the 4-lane group
                float mc = val * vpf;
                mc += __shfl_xor(mc, 1, 32);
                mc += __shfl_xor(mc, 2, 32);
                if (f == 0)
                    out[((size_t)(b * Cc + c) * Hh + h) * Wd + ww] = mc;
            }
        }
    }
}

extern "C" void kernel_launch(void* const* d_in, const int* in_sizes, int n_in,
                              void* d_out, int out_size, void* d_ws, size_t ws_size,
                              hipStream_t stream) {
    const float* frames = (const float*)d_in[0];
    const float* tvec   = (const float*)d_in[1];
    const float* We     = (const float*)d_in[2];
    const float* Wm     = (const float*)d_in[3];
    const float* Wg     = (const float*)d_in[4];
    const float* bg     = (const float*)d_in[5];
    float* outp = (float*)d_out;
    (void)in_sizes; (void)n_in; (void)out_size; (void)d_ws; (void)ws_size;

    dim3 grid(Bb * Hh);   // one block per (batch, image row)
    dim3 block(128);      // 4 waves, each owns a 16-pixel WMMA tile
    hipLaunchKernelGGL(vinr_fused, grid, block, 0, stream,
                       frames, tvec, We, Wm, Wg, bg, outp);
}
